// DiTBlock_31963146617108
// MI455X (gfx1250) — matrix operand order repaired and checked by reading, not compiled, not run on hardware
//
#include <hip/hip_runtime.h>
#include <hip/hip_bf16.h>

// ---------------- problem dims ----------------
constexpr int B_ = 2, L_ = 2048, D_ = 768, H_ = 12, HD_ = 64, DH_ = 3072;
constexpr int NTOK_ = B_ * L_;                 // 4096
constexpr float INV_SCALE_ = 0.125f;           // 1/sqrt(64)
constexpr float EPS_ = 1e-6f;

typedef __bf16 bf16;
typedef __attribute__((ext_vector_type(16))) __bf16 v16bf;
typedef __attribute__((ext_vector_type(8)))  __bf16 v8bf;
typedef __attribute__((ext_vector_type(8)))  float  v8f;

// ---------------- WMMA helpers ----------------
__device__ __forceinline__ v8f wmma_bf(v16bf a, v16bf b, v8f c) {
  // v_wmma_f32_16x16x32_bf16  (8-arg form)
  return __builtin_amdgcn_wmma_f32_16x16x32_bf16(false, a, false, b, (short)0, c,
                                                 false, false);
}

__device__ __forceinline__ v16bf make16(v8bf lo, v8bf hi) {
  return __builtin_shufflevector(lo, hi, 0, 1, 2, 3, 4, 5, 6, 7, 8, 9, 10, 11,
                                 12, 13, 14, 15);
}

// A fragment (16x32 bf16, row-major source). lane<16: row=row0+lane, K={0..7,16..23};
// lane>=16: same row, K={8..15,24..31}   (ISA 7.12.2 16-bit A layout)
__device__ __forceinline__ v16bf load_afrag_g(const bf16* base, int ld, int row0,
                                              int k0) {
  int lane = threadIdx.x & 31;
  int hf = lane >> 4;
  const bf16* p = base + (size_t)(row0 + (lane & 15)) * ld + k0 + hf * 8;
  v8bf lo = *(const v8bf*)(p);
  v8bf hi = *(const v8bf*)(p + 16);
  return make16(lo, hi);
}

// A fragment from a 16x32 LDS tile (ld = 32)
__device__ __forceinline__ v16bf load_afrag_lds(const bf16* base) {
  int lane = threadIdx.x & 31;
  int hf = lane >> 4;
  const bf16* p = base + (lane & 15) * 32 + hf * 8;
  v8bf lo = *(const v8bf*)(p);
  v8bf hi = *(const v8bf*)(p + 16);
  return make16(lo, hi);
}

// B fragment from B^T tile (N x K row-major, LDS or global):
// lane holds column n0+(lane&15); K = k0 + (lane>>4)*8 + {0..7,16..23} (matches A layout)
__device__ __forceinline__ v16bf load_bfrag(const bf16* baseT, int ld, int n0,
                                            int k0) {
  int lane = threadIdx.x & 31;
  const bf16* p = baseT + (size_t)(n0 + (lane & 15)) * ld + k0 + (lane >> 4) * 8;
  v8bf lo = *(const v8bf*)(p);
  v8bf hi = *(const v8bf*)(p + 16);
  return make16(lo, hi);
}

// ---------------- CDNA5 async memory->LDS copy ----------------
// Each active lane copies 16B: LDS[vdst_lane] = MEM[vaddr_lane].  Tracked by ASYNCcnt.
__device__ __forceinline__ void async_copy_b128(unsigned lds_off, const void* gaddr) {
  asm volatile("global_load_async_to_lds_b128 %0, %1, off"
               :
               : "v"(lds_off), "v"((unsigned long long)(uintptr_t)gaddr)
               : "memory");
}
__device__ __forceinline__ void wait_async0() {
  asm volatile("s_wait_asynccnt 0" ::: "memory");
}
__device__ __forceinline__ unsigned lds_addr32(const void* p) {
  return (unsigned)(uintptr_t)p;   // generic LDS addr: low 32 bits = LDS byte offset
}

// ---------------- small elementwise kernels ----------------
__global__ void cond_hidden_kernel(const float* c, const float* w1, const float* b1,
                                   float* hid) {
  int tid = blockIdx.x * 256 + threadIdx.x;
  if (tid >= B_ * DH_) return;
  int b = tid / DH_, j = tid % DH_;
  float acc = b1[j];
  for (int k = 0; k < D_; ++k) acc += c[b * D_ + k] * w1[(size_t)k * DH_ + j];
  hid[tid] = acc / (1.f + __expf(-acc));   // silu
}

__global__ void cond_out_kernel(const float* hid, const float* w2, const float* b2,
                                float* c2) {
  int tid = blockIdx.x * 256 + threadIdx.x;
  if (tid >= B_ * D_) return;
  int b = tid / D_, n = tid % D_;
  float acc = b2[n];
  for (int j = 0; j < DH_; ++j) acc += hid[b * DH_ + j] * w2[(size_t)j * D_ + n];
  c2[tid] = acc;
}

// mods layout: [slot][B][D]; slots: 0=alpha 1=beta_attn 2=gamma_attn 3=beta_ffn 4=gamma_ffn
__global__ void mods_kernel(const float* c2, const float* w0, const float* w1,
                            const float* w2, const float* w3, const float* w4,
                            float* mods) {
  int tid = blockIdx.x * 256 + threadIdx.x;
  if (tid >= 5 * B_ * D_) return;
  int slot = tid / (B_ * D_);
  int b = (tid / D_) % B_;
  int n = tid % D_;
  const float* W = slot == 0 ? w0 : slot == 1 ? w1 : slot == 2 ? w2 : slot == 3 ? w3 : w4;
  float acc = 0.f;
  for (int k = 0; k < D_; ++k) acc += c2[b * D_ + k] * W[(size_t)k * D_ + n];
  mods[tid] = acc;
}

// W (K x N fp32) -> Wt (N x K bf16)
__global__ void transpose_kernel(const float* W, unsigned short* Wt_, int K, int N) {
  int tid = blockIdx.x * 256 + threadIdx.x;
  if (tid >= K * N) return;
  int n = tid / K, k = tid % K;
  ((bf16*)Wt_)[tid] = (bf16)W[(size_t)k * N + n];
}

// h = bf16( normw * x * rsqrt(mean(x^2)+eps) * modg[b] + modb[b] )
__global__ void rmsnorm_mod_kernel(const float* x, const float* nw, const float* modg,
                                   const float* modb, unsigned short* out_) {
  __shared__ float red[256];
  int t = blockIdx.x;          // token
  int b = t / L_;
  int tid = threadIdx.x;
  const float* xr = x + (size_t)t * D_;
  float v0 = xr[tid], v1 = xr[tid + 256], v2 = xr[tid + 512];
  red[tid] = v0 * v0 + v1 * v1 + v2 * v2;
  __syncthreads();
  for (int s = 128; s > 0; s >>= 1) {
    if (tid < s) red[tid] += red[tid + s];
    __syncthreads();
  }
  float scale = rsqrtf(red[0] / (float)D_ + EPS_);
  bf16* out = (bf16*)out_;
  const float* mg = modg + (size_t)b * D_;
  const float* mb = modb + (size_t)b * D_;
#pragma unroll
  for (int j = 0; j < 3; ++j) {
    int i = tid + j * 256;
    out[(size_t)t * D_ + i] = (bf16)(nw[i] * xr[i] * scale * mg[i] + mb[i]);
  }
}

// RoPE + pack (b,l,h*64+d) fp32 -> (b,h,l,d) bf16 for Q and K
__global__ void rope_pack_kernel(const float* Qf, const float* Kf, const float* cs,
                                 const float* sn, unsigned short* Qb_,
                                 unsigned short* Kb_) {
  int tid = blockIdx.x * 256 + threadIdx.x;
  if (tid >= B_ * H_ * L_ * 32) return;
  int i = tid & 31;
  int l = (tid >> 5) & (L_ - 1);
  int h = (tid / (32 * L_)) % H_;
  int b = tid / (32 * L_ * H_);
  float c = cs[l * 32 + i], s = sn[l * 32 + i];
  size_t ib = (size_t)(b * L_ + l) * D_ + h * HD_;
  size_t ob = ((size_t)(b * H_ + h) * L_ + l) * HD_;
  bf16* Qb = (bf16*)Qb_;
  bf16* Kb = (bf16*)Kb_;
  float q0 = Qf[ib + 2 * i], q1 = Qf[ib + 2 * i + 1];
  Qb[ob + 2 * i]     = (bf16)(q0 * c - q1 * s);
  Qb[ob + 2 * i + 1] = (bf16)(q0 * s + q1 * c);
  float k0 = Kf[ib + 2 * i], k1 = Kf[ib + 2 * i + 1];
  Kb[ob + 2 * i]     = (bf16)(k0 * c - k1 * s);
  Kb[ob + 2 * i + 1] = (bf16)(k0 * s + k1 * c);
}

// V (b,l,h*64+d) fp32 -> V^T (b,h,d,l) bf16
__global__ void vpack_kernel(const float* Vf, unsigned short* Vt_) {
  int tid = blockIdx.x * 256 + threadIdx.x;
  if (tid >= B_ * H_ * L_ * HD_) return;
  int d = tid & 63;
  int l = (tid >> 6) & (L_ - 1);
  int h = (tid / (HD_ * L_)) % H_;
  int b = tid / (HD_ * L_ * H_);
  ((bf16*)Vt_)[((size_t)(b * H_ + h) * HD_ + d) * L_ + l] =
      (bf16)Vf[(size_t)(b * L_ + l) * D_ + h * HD_ + d];
}

// out = xin + y * mod[b][n]
__global__ void resid_kernel(const float* xin, const float* y, const float* mod,
                             float* out) {
  int tid = blockIdx.x * 256 + threadIdx.x;
  if (tid >= NTOK_ * D_) return;
  int n = tid % D_;
  int b = (tid / D_) / L_;
  out[tid] = xin[tid] + y[tid] * mod[b * D_ + n];
}

// ---------------- WMMA GEMM: C(MxN) = A(MxK bf16) * Bt(NxK bf16)^T ----------------
// B tile (64 cols x 32 K = 4KB) is double-buffered in LDS via async copy:
// each wave issues one global_load_async_to_lds_b128 (512B) per k-step for tile i+1,
// then computes tile i.  One block barrier per k-step.
// mode 0: C fp32.  mode 1: Cbf = bf16( silu(gatef) * acc )   (fused SwiGLU epilogue)
__global__ void gemm_bf16_kernel(const unsigned short* A_, const unsigned short* Bt_,
                                 float* C, unsigned short* Cbf_, const float* gatef,
                                 int M, int N, int K, int mode) {
  __shared__ __align__(16) unsigned short bt_raw[2][64 * 32];   // 2 x 4KB
  const bf16* A = (const bf16*)A_;
  const bf16* Bt = (const bf16*)Bt_;
  int wave = threadIdx.x >> 5;
  int lane = threadIdx.x & 31;
  int row0 = blockIdx.y * 128 + wave * 16;
  int n0 = blockIdx.x * 64;

  // async-copy coordinates: wave w covers tile rows w*8..w*8+7; 4 lanes x 16B per row
  int crow = wave * 8 + (lane >> 2);
  int cbyte = (lane & 3) * 16;
  const char* gsrc = (const char*)(Bt + (size_t)(n0 + crow) * K) + cbyte;
  unsigned ldsb[2];
  ldsb[0] = lds_addr32(&bt_raw[0][0]) + crow * 64 + cbyte;
  ldsb[1] = lds_addr32(&bt_raw[1][0]) + crow * 64 + cbyte;

  const bf16* arow = A + (size_t)(row0 + (lane & 15)) * K;  // for prefetch

  int nk = K / 32;
  async_copy_b128(ldsb[0], gsrc);                // tile 0 -> buf 0
  v8f acc[4] = {};
  for (int i = 0; i < nk; ++i) {
    int k0 = i * 32;
    wait_async0();        // my chunk of tile i landed
    __syncthreads();      // everyone's chunk landed; buf (i+1)&1 free to overwrite
    if (i + 1 < nk)
      async_copy_b128(ldsb[(i + 1) & 1], gsrc + (size_t)(k0 + 32) * 2);
    if (k0 + 32 < K) __builtin_prefetch(arow + k0 + 32, 0, 3);  // global_prefetch_b8
    const bf16* btile = (const bf16*)bt_raw[i & 1];
    v16bf a = load_afrag_g(A, K, row0, k0);
#pragma unroll
    for (int t = 0; t < 4; ++t) {
      v16bf b = load_bfrag(btile, 32, t * 16, 0);
      acc[t] = wmma_bf(a, b, acc[t]);
    }
  }
  int hf = lane >> 4;
#pragma unroll
  for (int t = 0; t < 4; ++t) {
#pragma unroll
    for (int v = 0; v < 8; ++v) {
      int m = row0 + v + 8 * hf;
      int n = n0 + t * 16 + (lane & 15);
      size_t idx = (size_t)m * N + n;
      if (mode == 0) {
        C[idx] = acc[t][v];
      } else {
        float g = gatef[idx];
        float sg = g / (1.f + __expf(-g));
        ((bf16*)Cbf_)[idx] = (bf16)(sg * acc[t][v]);
      }
    }
  }
}

// ---------------- flash attention ----------------
// Block = 8 waves = 128 query rows of one (b,h).  K/V tiles (32 keys: 4KB+4KB)
// double-buffered in LDS via async copies (one K + one V instruction per wave per
// tile); each wave runs online softmax over its 16 query rows.
__global__ void attn_kernel(const unsigned short* Qb_, const unsigned short* Kb_,
                            const unsigned short* Vt_, unsigned short* Oa_) {
  __shared__ __align__(16) unsigned short kt_raw[2][32 * 64];   // 2 x 4KB
  __shared__ __align__(16) unsigned short vt_raw[2][64 * 32];   // 2 x 4KB
  __shared__ __align__(32) unsigned short pt_raw[8][512];       // per-wave P tile
  const bf16* Qb = (const bf16*)Qb_;
  const bf16* Kb = (const bf16*)Kb_;
  int wave = threadIdx.x >> 5;
  int lane = threadIdx.x & 31;
  int hf = lane >> 4;
  constexpr int QBLK = L_ / 128;           // 16 q-blocks per (b,h)
  int bh = blockIdx.x / QBLK;
  int qblk = blockIdx.x % QBLK;
  int b = bh / H_, h = bh % H_;
  int q0 = qblk * 128 + wave * 16;
  const bf16* Qh = Qb + (size_t)bh * L_ * HD_;
  const bf16* Kh = Kb + (size_t)bh * L_ * HD_;
  const bf16* Vh = (const bf16*)Vt_ + (size_t)bh * HD_ * L_;
  bf16* myp = (bf16*)pt_raw[wave];

  // async-copy coordinates
  int krow = wave * 4 + (lane >> 3);        // 0..31 key row (128B each)
  int kbyte = (lane & 7) * 16;
  int vrow = wave * 8 + (lane >> 2);        // 0..63 d row (64B each)
  int vbyte = (lane & 3) * 16;
  const char* gK = (const char*)(Kh + (size_t)krow * HD_) + kbyte;
  const char* gV = (const char*)(Vh + (size_t)vrow * L_) + vbyte;
  unsigned ldsk[2], ldsv[2];
  ldsk[0] = lds_addr32(&kt_raw[0][0]) + krow * 128 + kbyte;
  ldsk[1] = lds_addr32(&kt_raw[1][0]) + krow * 128 + kbyte;
  ldsv[0] = lds_addr32(&vt_raw[0][0]) + vrow * 64 + vbyte;
  ldsv[1] = lds_addr32(&vt_raw[1][0]) + vrow * 64 + vbyte;

  v16bf qf0 = load_afrag_g(Qh, HD_, q0, 0);
  v16bf qf1 = load_afrag_g(Qh, HD_, q0, 32);

  v8f o[4] = {};
  float mi[8], li[8];
#pragma unroll
  for (int v = 0; v < 8; ++v) { mi[v] = -1e30f; li[v] = 0.f; }

  constexpr int NIT = L_ / 32;
  // prologue: tile 0 -> buf 0
  async_copy_b128(ldsk[0], gK);
  async_copy_b128(ldsv[0], gV);
  for (int it = 0; it < NIT; ++it) {
    int m0 = it * 32;
    int p = it & 1;
    wait_async0();        // my chunks of tile it landed
    __syncthreads();      // all chunks landed; other buffer free to overwrite
    if (it + 1 < NIT) {
      async_copy_b128(ldsk[1 - p], gK + (size_t)(m0 + 32) * HD_ * 2);
      async_copy_b128(ldsv[1 - p], gV + (size_t)(m0 + 32) * 2);
    }
    const bf16* ktile = (const bf16*)kt_raw[p];
    const bf16* vtile = (const bf16*)vt_raw[p];

    v16bf kf00 = load_bfrag(ktile, 64, 0, 0);
    v16bf kf01 = load_bfrag(ktile, 64, 0, 32);
    v16bf kf10 = load_bfrag(ktile, 64, 16, 0);
    v16bf kf11 = load_bfrag(ktile, 64, 16, 32);
    v8f z = {};
    v8f s0 = wmma_bf(qf0, kf00, z);
    s0 = wmma_bf(qf1, kf01, s0);
    v8f s1 = wmma_bf(qf0, kf10, z);
    s1 = wmma_bf(qf1, kf11, s1);

    float sf[8];
#pragma unroll
    for (int v = 0; v < 8; ++v) {
      float a = s0[v] * INV_SCALE_;
      float c = s1[v] * INV_SCALE_;
      float mx = fmaxf(a, c);
      mx = fmaxf(mx, __shfl_xor(mx, 1, 32));
      mx = fmaxf(mx, __shfl_xor(mx, 2, 32));
      mx = fmaxf(mx, __shfl_xor(mx, 4, 32));
      mx = fmaxf(mx, __shfl_xor(mx, 8, 32));
      float mnew = fmaxf(mi[v], mx);
      sf[v] = __expf(mi[v] - mnew);
      float p0 = __expf(a - mnew);
      float p1 = __expf(c - mnew);
      float rs = p0 + p1;
      rs += __shfl_xor(rs, 1, 32);
      rs += __shfl_xor(rs, 2, 32);
      rs += __shfl_xor(rs, 4, 32);
      rs += __shfl_xor(rs, 8, 32);
      li[v] = li[v] * sf[v] + rs;
      mi[v] = mnew;
      int r = v + 8 * hf;                       // C-frag row -> P tile row
      myp[r * 32 + (lane & 15)] = (bf16)p0;
      myp[r * 32 + 16 + (lane & 15)] = (bf16)p1;
    }
#pragma unroll
    for (int t = 0; t < 4; ++t)
#pragma unroll
      for (int v = 0; v < 8; ++v) o[t][v] *= sf[v];

    asm volatile("s_wait_dscnt 0" ::: "memory");   // P stores visible before reload
    v16bf pf = load_afrag_lds(myp);
#pragma unroll
    for (int t = 0; t < 4; ++t) {
      v16bf vf = load_bfrag(vtile, 32, t * 16, 0);
      o[t] = wmma_bf(pf, vf, o[t]);
    }
  }

  bf16* Oa = (bf16*)Oa_;
#pragma unroll
  for (int t = 0; t < 4; ++t)
#pragma unroll
    for (int v = 0; v < 8; ++v) {
      int l = q0 + v + 8 * hf;
      int dd = t * 16 + (lane & 15);
      Oa[(size_t)(b * L_ + l) * D_ + h * HD_ + dd] = (bf16)(o[t][v] / li[v]);
    }
}

// ---------------- workspace layout ----------------
constexpr size_t alup(size_t x) { return (x + 255) & ~(size_t)255; }
constexpr size_t SZ_CONDHID = alup((size_t)B_ * DH_ * 4);
constexpr size_t SZ_C2 = alup((size_t)B_ * D_ * 4);
constexpr size_t SZ_MODS = alup((size_t)5 * B_ * D_ * 4);
constexpr size_t SZ_TOKD_F = alup((size_t)NTOK_ * D_ * 4);
constexpr size_t SZ_TOKDH_F = alup((size_t)NTOK_ * DH_ * 4);
constexpr size_t SZ_DD_B = alup((size_t)D_ * D_ * 2);
constexpr size_t SZ_DDH_B = alup((size_t)D_ * DH_ * 2);
constexpr size_t SZ_TOKD_B = alup((size_t)NTOK_ * D_ * 2);
constexpr size_t SZ_TOKDH_B = alup((size_t)NTOK_ * DH_ * 2);

constexpr size_t OFF_CONDHID = 0;
constexpr size_t OFF_C2 = OFF_CONDHID + SZ_CONDHID;
constexpr size_t OFF_MODS = OFF_C2 + SZ_C2;
constexpr size_t OFF_QF = OFF_MODS + SZ_MODS;
constexpr size_t OFF_KF = OFF_QF + SZ_TOKD_F;
constexpr size_t OFF_VF = OFF_KF + SZ_TOKD_F;
constexpr size_t OFF_APROJ = OFF_VF + SZ_TOKD_F;
constexpr size_t OFF_X2 = OFF_APROJ + SZ_TOKD_F;
constexpr size_t OFF_GATEF = OFF_X2 + SZ_TOKD_F;
constexpr size_t OFF_FFF = OFF_GATEF + SZ_TOKDH_F;
constexpr size_t OFF_WQT = OFF_FFF + SZ_TOKD_F;
constexpr size_t OFF_WKT = OFF_WQT + SZ_DD_B;
constexpr size_t OFF_WVT = OFF_WKT + SZ_DD_B;
constexpr size_t OFF_WOT = OFF_WVT + SZ_DD_B;
constexpr size_t OFF_GATET = OFF_WOT + SZ_DD_B;
constexpr size_t OFF_HIDT = OFF_GATET + SZ_DDH_B;
constexpr size_t OFF_OUTT = OFF_HIDT + SZ_DDH_B;
constexpr size_t OFF_H1 = OFF_OUTT + SZ_DDH_B;
constexpr size_t OFF_QB = OFF_H1 + SZ_TOKD_B;
constexpr size_t OFF_KB = OFF_QB + SZ_TOKD_B;
constexpr size_t OFF_VT = OFF_KB + SZ_TOKD_B;
constexpr size_t OFF_OA = OFF_VT + SZ_TOKD_B;
constexpr size_t OFF_H2 = OFF_OA + SZ_TOKD_B;
constexpr size_t OFF_FFIN = OFF_H2 + SZ_TOKD_B;

extern "C" void kernel_launch(void* const* d_in, const int* in_sizes, int n_in,
                              void* d_out, int out_size, void* d_ws, size_t ws_size,
                              hipStream_t stream) {
  const float* x = (const float*)d_in[0];
  const float* c = (const float*)d_in[1];
  const float* fcos = (const float*)d_in[2];
  const float* fsin = (const float*)d_in[3];
  const float* W_q = (const float*)d_in[4];
  const float* W_k = (const float*)d_in[5];
  const float* W_v = (const float*)d_in[6];
  const float* W_o = (const float*)d_in[7];
  const float* cond_w1 = (const float*)d_in[8];
  const float* cond_b1 = (const float*)d_in[9];
  const float* cond_w2 = (const float*)d_in[10];
  const float* cond_b2 = (const float*)d_in[11];
  const float* attn_norm_w = (const float*)d_in[12];
  const float* attn_alpha = (const float*)d_in[13];
  const float* attn_beta = (const float*)d_in[14];
  const float* attn_gamma = (const float*)d_in[15];
  const float* ffn_gate = (const float*)d_in[16];
  const float* ffn_hidden = (const float*)d_in[17];
  const float* ffn_out = (const float*)d_in[18];
  const float* ffn_norm_w = (const float*)d_in[19];
  const float* ffn_beta = (const float*)d_in[20];
  const float* ffn_gamma = (const float*)d_in[21];
  float* out = (float*)d_out;

  char* w = (char*)d_ws;
  float* condHid = (float*)(w + OFF_CONDHID);
  float* c2 = (float*)(w + OFF_C2);
  float* mods = (float*)(w + OFF_MODS);
  float* Qf = (float*)(w + OFF_QF);
  float* Kf = (float*)(w + OFF_KF);
  float* Vf = (float*)(w + OFF_VF);
  float* aproj = (float*)(w + OFF_APROJ);
  float* x2 = (float*)(w + OFF_X2);
  float* gatef = (float*)(w + OFF_GATEF);
  float* fff = (float*)(w + OFF_FFF);
  unsigned short* WqT = (unsigned short*)(w + OFF_WQT);
  unsigned short* WkT = (unsigned short*)(w + OFF_WKT);
  unsigned short* WvT = (unsigned short*)(w + OFF_WVT);
  unsigned short* WoT = (unsigned short*)(w + OFF_WOT);
  unsigned short* gateT = (unsigned short*)(w + OFF_GATET);
  unsigned short* hidT = (unsigned short*)(w + OFF_HIDT);
  unsigned short* outT = (unsigned short*)(w + OFF_OUTT);
  unsigned short* h1 = (unsigned short*)(w + OFF_H1);
  unsigned short* Qb = (unsigned short*)(w + OFF_QB);
  unsigned short* Kb = (unsigned short*)(w + OFF_KB);
  unsigned short* Vt = (unsigned short*)(w + OFF_VT);
  unsigned short* Oa = (unsigned short*)(w + OFF_OA);
  unsigned short* h2 = (unsigned short*)(w + OFF_H2);
  unsigned short* ffin = (unsigned short*)(w + OFF_FFIN);

  const float* modAlpha = mods + 0 * B_ * D_;
  const float* modBetaA = mods + 1 * B_ * D_;
  const float* modGamA = mods + 2 * B_ * D_;
  const float* modBetaF = mods + 3 * B_ * D_;
  const float* modGamF = mods + 4 * B_ * D_;

  // 1) conditioning MLP + modulation vectors
  cond_hidden_kernel<<<(B_ * DH_ + 255) / 256, 256, 0, stream>>>(c, cond_w1, cond_b1, condHid);
  cond_out_kernel<<<(B_ * D_ + 255) / 256, 256, 0, stream>>>(condHid, cond_w2, cond_b2, c2);
  mods_kernel<<<(5 * B_ * D_ + 255) / 256, 256, 0, stream>>>(c2, attn_alpha, attn_beta,
                                                             attn_gamma, ffn_beta,
                                                             ffn_gamma, mods);
  // 2) weights -> bf16 transposed (N x K)
  transpose_kernel<<<(D_ * D_ + 255) / 256, 256, 0, stream>>>(W_q, WqT, D_, D_);
  transpose_kernel<<<(D_ * D_ + 255) / 256, 256, 0, stream>>>(W_k, WkT, D_, D_);
  transpose_kernel<<<(D_ * D_ + 255) / 256, 256, 0, stream>>>(W_v, WvT, D_, D_);
  transpose_kernel<<<(D_ * D_ + 255) / 256, 256, 0, stream>>>(W_o, WoT, D_, D_);
  transpose_kernel<<<(D_ * DH_ + 255) / 256, 256, 0, stream>>>(ffn_gate, gateT, D_, DH_);
  transpose_kernel<<<(D_ * DH_ + 255) / 256, 256, 0, stream>>>(ffn_hidden, hidT, D_, DH_);
  transpose_kernel<<<(DH_ * D_ + 255) / 256, 256, 0, stream>>>(ffn_out, outT, DH_, D_);

  // 3) h1 = modulated rmsnorm(x)
  rmsnorm_mod_kernel<<<NTOK_, 256, 0, stream>>>(x, attn_norm_w, modGamA, modBetaA, h1);

  // 4) QKV projections (WMMA)
  dim3 gD(D_ / 64, NTOK_ / 128);
  gemm_bf16_kernel<<<gD, 256, 0, stream>>>(h1, WqT, Qf, nullptr, nullptr, NTOK_, D_, D_, 0);
  gemm_bf16_kernel<<<gD, 256, 0, stream>>>(h1, WkT, Kf, nullptr, nullptr, NTOK_, D_, D_, 0);
  gemm_bf16_kernel<<<gD, 256, 0, stream>>>(h1, WvT, Vf, nullptr, nullptr, NTOK_, D_, D_, 0);

  // 5) RoPE + pack to per-head layouts
  rope_pack_kernel<<<(B_ * H_ * L_ * 32 + 255) / 256, 256, 0, stream>>>(Qf, Kf, fcos,
                                                                        fsin, Qb, Kb);
  vpack_kernel<<<(B_ * H_ * L_ * HD_ + 255) / 256, 256, 0, stream>>>(Vf, Vt);

  // 6) flash attention (WMMA, async-LDS double-buffered K/V)
  attn_kernel<<<B_ * H_ * (L_ / 128), 256, 0, stream>>>(Qb, Kb, Vt, Oa);

  // 7) O projection + gated residual
  gemm_bf16_kernel<<<gD, 256, 0, stream>>>(Oa, WoT, aproj, nullptr, nullptr, NTOK_, D_, D_, 0);
  resid_kernel<<<(NTOK_ * D_ + 255) / 256, 256, 0, stream>>>(x, aproj, modAlpha, x2);

  // 8) FFN: modulated rmsnorm, gate/hidden GEMMs (fused silu*mul), down proj
  rmsnorm_mod_kernel<<<NTOK_, 256, 0, stream>>>(x2, ffn_norm_w, modGamF, modBetaF, h2);
  dim3 gDH(DH_ / 64, NTOK_ / 128);
  gemm_bf16_kernel<<<gDH, 256, 0, stream>>>(h2, gateT, gatef, nullptr, nullptr, NTOK_, DH_, D_, 0);
  gemm_bf16_kernel<<<gDH, 256, 0, stream>>>(h2, hidT, nullptr, ffin, gatef, NTOK_, DH_, D_, 1);
  gemm_bf16_kernel<<<gD, 256, 0, stream>>>(ffin, outT, fff, nullptr, nullptr, NTOK_, D_, DH_, 0);

  // 9) final gated residual (reference multiplies by mod(ffn_gamma) again)
  resid_kernel<<<(NTOK_ * D_ + 255) / 256, 256, 0, stream>>>(x2, fff, modGamF, out);
}
